// TransformerEncoder__62457414418602
// MI455X (gfx1250) — matrix operand order repaired and checked by reading, not compiled, not run on hardware
//
#include <hip/hip_runtime.h>
#include <hip/hip_bf16.h>
#include <math.h>

// ---------------- types / helpers -------------------------------------------
typedef __attribute__((ext_vector_type(16))) __bf16 v16bf;
typedef __attribute__((ext_vector_type(8)))  float  v8f;
typedef unsigned short u16;

__device__ __forceinline__ u16 f2bf(float f) {
    unsigned u = __float_as_uint(f);
    u += 0x7fffu + ((u >> 16) & 1u);          // round-to-nearest-even
    return (u16)(u >> 16);
}

__device__ __forceinline__ v8f vzero8() {
    v8f z;
#pragma unroll
    for (int i = 0; i < 8; ++i) z[i] = 0.0f;
    return z;
}

// load a 16-element bf16 WMMA fragment as two 16B chunks (ds_load_b128)
__device__ __forceinline__ v16bf ld_frag(const u16* p0, const u16* p1) {
    union { uint4 q[2]; v16bf v; } t;
    t.q[0] = *(const uint4*)p0;
    t.q[1] = *(const uint4*)p1;
    return t.v;
}

__device__ __forceinline__ v8f wmma_bf16(v16bf a, v16bf b, v8f c) {
    return __builtin_amdgcn_wmma_f32_16x16x32_bf16(false, a, false, b, (short)0, c, false, false);
}

// ---- CDNA5 async global->LDS copy (ASYNCcnt) --------------------------------
// GVS form: LDS byte address in VGPR, 32-bit unsigned byte offset in VGPR,
// 64-bit base in SGPR pair. Flat shared pointers: low 32 bits == LDS offset.
__device__ __forceinline__ unsigned lds_off(const void* p) {
    return (unsigned)(unsigned long long)p;
}
__device__ __forceinline__ void async_ld_b128(unsigned ldsaddr, const void* gbase,
                                              unsigned byte_off) {
    asm volatile("global_load_async_to_lds_b128 %0, %1, %2"
                 :: "v"(ldsaddr), "v"(byte_off), "s"(gbase) : "memory");
}
__device__ __forceinline__ void wait_async0() {
    asm volatile("s_wait_asynccnt 0x0" ::: "memory");
}

// ---------------- elementwise conversions ------------------------------------
__global__ void cvt_k(const float* __restrict__ in, u16* __restrict__ out, int n) {
    int i = blockIdx.x * 256 + threadIdx.x;
    if (i < n) out[i] = f2bf(in[i]);
}

// W1 = knn_w[0:384]; Wd = knn_w[384:768] - knn_w[0:384]   (row-major 384x384 each)
__global__ void cvt_wd_k(const float* __restrict__ knnw, u16* __restrict__ w1,
                         u16* __restrict__ wd, int n) {
    int i = blockIdx.x * 256 + threadIdx.x;
    if (i < n) {
        float a = knnw[i];
        w1[i] = f2bf(a);
        wd[i] = f2bf(knnw[n + i] - a);
    }
}

// ---------------- layernorm -> bf16 ------------------------------------------
__global__ __launch_bounds__(128)
void ln_k(const float* __restrict__ in, const float* __restrict__ w,
          const float* __restrict__ bia, u16* __restrict__ out, int D) {
    int row = blockIdx.x, tid = threadIdx.x;
    __shared__ float r1[128], r2[128];
    float s = 0.f, ss = 0.f;
    for (int c = tid; c < D; c += 128) { float x = in[row * D + c]; s += x; ss += x * x; }
    r1[tid] = s; r2[tid] = ss; __syncthreads();
    for (int off = 64; off > 0; off >>= 1) {
        if (tid < off) { r1[tid] += r1[tid + off]; r2[tid] += r2[tid + off]; }
        __syncthreads();
    }
    float mu = r1[0] / D;
    float var = r2[0] / D - mu * mu;
    float rs = rsqrtf(var + 1e-5f);
    for (int c = tid; c < D; c += 128) {
        float x = (in[row * D + c] - mu) * rs * w[c] + bia[c];
        out[row * D + c] = f2bf(x);
    }
}

// ---------------- generic bf16 WMMA GEMM (128x64 tile, 8 waves) ---------------
// MODE 0: f32 out (+bias, +residual)      MODE 1: bf16 out (+bias, act==1 -> GELU)
// MODE 2: QKV scatter bf16 [3][8][6][2048][64], Q scaled by 0.125
template <int MODE>
__global__ __launch_bounds__(256)
void gemm_bf16_k(const u16* __restrict__ A, int lda,
                 const u16* __restrict__ W, int ldw,
                 int M, int N, int K,
                 const float* __restrict__ bias,
                 const float* __restrict__ resid, int ldr,
                 float* __restrict__ outF, u16* __restrict__ outB, int ldo, int act) {
    __shared__ __align__(16) u16 sA[128 * 40];
    __shared__ __align__(16) u16 sB[64 * 40];   // transposed: [n][k]
    const int tid = threadIdx.x, wv = tid >> 5, lane = tid & 31;
    const int hg = lane >> 4, ln = lane & 15;
    const int m0 = blockIdx.y * 128, n0 = blockIdx.x * 64;

    v8f acc[4];
#pragma unroll
    for (int nt = 0; nt < 4; ++nt) acc[nt] = vzero8();

    for (int k0 = 0; k0 < K; k0 += 32) {
        if (k0 + 32 < K)
            __builtin_prefetch(&A[(size_t)(m0 + (tid & 127)) * lda + k0 + 32], 0, 1);
        // async copy A tile 128x32 (16B chunks)
        for (int ch = tid; ch < 128 * 4; ch += 256) {
            int r = ch >> 2, c8 = (ch & 3) * 8;
            async_ld_b128(lds_off(&sA[r * 40 + c8]), A,
                          (unsigned)(((size_t)(m0 + r) * lda + k0 + c8) * 2));
        }
        // B tile transposed (scalar staging)
        for (int i = tid; i < 32 * 64; i += 256) {
            int k = i >> 6, n = i & 63;
            sB[n * 40 + k] = W[(size_t)(k0 + k) * ldw + n0 + n];
        }
        wait_async0();
        __syncthreads();
        v16bf af = ld_frag(sA + (wv * 16 + ln) * 40 + hg * 8,
                           sA + (wv * 16 + ln) * 40 + 16 + hg * 8);
#pragma unroll
        for (int nt = 0; nt < 4; ++nt) {
            v16bf bf = ld_frag(sB + (nt * 16 + ln) * 40 + hg * 8,
                               sB + (nt * 16 + ln) * 40 + 16 + hg * 8);
            acc[nt] = wmma_bf16(af, bf, acc[nt]);
        }
        __syncthreads();
    }

#pragma unroll
    for (int nt = 0; nt < 4; ++nt) {
#pragma unroll
        for (int v = 0; v < 8; ++v) {
            int row = m0 + wv * 16 + hg * 8 + v;
            int col = n0 + nt * 16 + ln;
            if (row >= M || col >= N) continue;
            float x = acc[nt][v];
            if (bias) x += bias[col];
            if (MODE == 0) {
                if (resid) x += resid[(size_t)row * ldr + col];
                outF[(size_t)row * ldo + col] = x;
            } else if (MODE == 1) {
                if (act == 1) x = 0.5f * x * (1.0f + erff(x * 0.70710678118654752f));
                outB[(size_t)row * ldo + col] = f2bf(x);
            } else {  // QKV scatter
                int b = row >> 11, n = row & 2047;
                int idx3 = col / 384, rem = col - idx3 * 384;
                int h = rem >> 6, d = rem & 63;
                if (idx3 == 0) x *= 0.125f;  // HEAD_DIM^-0.5
                outB[((size_t)((idx3 * 8 + b) * 6 + h) * 2048 + n) * 64 + d] = f2bf(x);
            }
        }
    }
}

// ---------------- flash attention (64 queries / block, one (b,h)) -------------
__global__ __launch_bounds__(128)
void attn_k(const u16* __restrict__ qkv, u16* __restrict__ outB) {
    const int tid = threadIdx.x, wv = tid >> 5, lane = tid & 31;
    const int hg = lane >> 4, ln = lane & 15;
    const int bid = blockIdx.x;
    const int qblk = bid & 31;            // 2048/64
    const int h = (bid >> 5) % 6;
    const int b = bid / 192;
    const u16* Q  = qkv + (size_t)((0 * 8 + b) * 6 + h) * 2048 * 64;
    const u16* Kg = qkv + (size_t)((1 * 8 + b) * 6 + h) * 2048 * 64;
    const u16* Vg = qkv + (size_t)((2 * 8 + b) * 6 + h) * 2048 * 64;

    __shared__ __align__(16) u16 sK[32 * 72];   // [key][dim]
    __shared__ __align__(16) u16 sV[64 * 40];   // transposed: [dim][key]

    const int gq = qblk * 64 + wv * 16 + ln;    // query whose stats this lane owns
    const u16* qrow = Q + (size_t)gq * 64;      // Q pre-scaled at QKV epilogue
    v16bf qf0 = ld_frag(qrow + hg * 8,      qrow + 16 + hg * 8);        // dims 0..31
    v16bf qf1 = ld_frag(qrow + 32 + hg * 8, qrow + 48 + hg * 8);   // dims 32..63

    v8f o[4];
#pragma unroll
    for (int nt = 0; nt < 4; ++nt) o[nt] = vzero8();
    float m = -3.0e38f, l = 0.0f;

    for (int kb = 0; kb < 2048; kb += 32) {
        // async copy K tile (32 keys x 64 dims, 16B chunks)
        for (int ch = tid; ch < 32 * 8; ch += 128) {
            int kk = ch >> 3, c8 = (ch & 7) * 8;
            async_ld_b128(lds_off(&sK[kk * 72 + c8]), Kg,
                          (unsigned)(((size_t)(kb + kk) * 64 + c8) * 2));
        }
        // V tile transposed (scalar staging)
        for (int i = tid; i < 2048; i += 128) {
            int kk = i >> 6, d = i & 63;
            sV[d * 40 + kk] = Vg[(size_t)(kb + kk) * 64 + d];
        }
        wait_async0();
        __syncthreads();

        // S^T tiles: rows = keys, cols = queries
        v8f s0 = vzero8(), s1 = vzero8();
        v16bf ka = ld_frag(sK + ln * 72 + hg * 8, sK + ln * 72 + 16 + hg * 8);
        s0 = wmma_bf16(ka, qf0, s0);
        ka = ld_frag(sK + ln * 72 + 32 + hg * 8, sK + ln * 72 + 48 + hg * 8);
        s0 = wmma_bf16(ka, qf1, s0);
        ka = ld_frag(sK + (16 + ln) * 72 + hg * 8, sK + (16 + ln) * 72 + 16 + hg * 8);
        s1 = wmma_bf16(ka, qf0, s1);
        ka = ld_frag(sK + (16 + ln) * 72 + 32 + hg * 8, sK + (16 + ln) * 72 + 48 + hg * 8);
        s1 = wmma_bf16(ka, qf1, s1);

        // online softmax over the 32 keys (per-query = per ln column)
        float tm = -3.0e38f;
#pragma unroll
        for (int v = 0; v < 8; ++v) tm = fmaxf(tm, fmaxf(s0[v], s1[v]));
        tm = fmaxf(tm, __shfl_xor(tm, 16, 32));
        float mnew = fmaxf(m, tm);
        float scale = __expf(m - mnew);
        float p0[8], p1[8], ts = 0.0f;
#pragma unroll
        for (int v = 0; v < 8; ++v) {
            p0[v] = __expf(s0[v] - mnew);
            p1[v] = __expf(s1[v] - mnew);
            ts += p0[v] + p1[v];
        }
        ts += __shfl_xor(ts, 16, 32);
        l = l * scale + ts;
        m = mnew;

        // pack P into an A-fragment (layout matches exactly)
        union { u16 s[16]; v16bf v; } pf;
#pragma unroll
        for (int v = 0; v < 8; ++v) { pf.s[v] = f2bf(p0[v]); pf.s[8 + v] = f2bf(p1[v]); }

        float sc[8];
#pragma unroll
        for (int v = 0; v < 8; ++v) sc[v] = __shfl(scale, hg * 8 + v, 16);
#pragma unroll
        for (int nt = 0; nt < 4; ++nt) {
            v8f t = o[nt];
#pragma unroll
            for (int v = 0; v < 8; ++v) t[v] *= sc[v];
            v16bf vf = ld_frag(sV + (nt * 16 + ln) * 40 + hg * 8,
                               sV + (nt * 16 + ln) * 40 + 16 + hg * 8);
            o[nt] = wmma_bf16(pf.v, vf, t);
        }
        __syncthreads();
    }

    float linv = 1.0f / l;
    float li[8];
#pragma unroll
    for (int v = 0; v < 8; ++v) li[v] = __shfl(linv, hg * 8 + v, 16);
#pragma unroll
    for (int nt = 0; nt < 4; ++nt)
#pragma unroll
        for (int v = 0; v < 8; ++v) {
            int row = b * 2048 + qblk * 64 + wv * 16 + hg * 8 + v;
            outB[(size_t)row * 384 + h * 64 + nt * 16 + ln] = f2bf(o[nt][v] * li[v]);
        }
}

// ---------------- KNN branch: max_k leakyrelu(gather(nx)@W1 + t2) -------------
__global__ __launch_bounds__(256)
void knn_k(const u16* __restrict__ nx, const u16* __restrict__ W1,
           const long long* __restrict__ knn_idx, const float* __restrict__ t2,
           u16* __restrict__ outB) {
    __shared__ __align__(16) u16 sA[128 * 40];
    __shared__ __align__(16) u16 sB[64 * 40];
    const int tid = threadIdx.x, wv = tid >> 5, lane = tid & 31;
    const int hg = lane >> 4, ln = lane & 15;
    const int m0 = blockIdx.y * 128, n0 = blockIdx.x * 64;

    float t2r[4][8];
    v8f best[4];
#pragma unroll
    for (int nt = 0; nt < 4; ++nt) {
#pragma unroll
        for (int v = 0; v < 8; ++v) {
            int row = m0 + wv * 16 + hg * 8 + v, col = n0 + nt * 16 + ln;
            t2r[nt][v] = t2[(size_t)row * 384 + col];
            best[nt][v] = -3.0e38f;
        }
    }

    for (int k = 0; k < 8; ++k) {
        v8f acc[4];
#pragma unroll
        for (int nt = 0; nt < 4; ++nt) acc[nt] = vzero8();
        for (int k0 = 0; k0 < 384; k0 += 32) {
            // async gather of A tile: 128 gathered rows x 32 cols (16B chunks)
            for (int ch = tid; ch < 128 * 4; ch += 256) {
                int r = ch >> 2, c8 = (ch & 3) * 8;
                int mrow = m0 + r, b = mrow >> 11, n = mrow & 2047;
                long long src = knn_idx[(long long)(b * 8 + k) * 2048 + n];
                async_ld_b128(lds_off(&sA[r * 40 + c8]), nx,
                              (unsigned)((src * 384 + k0 + c8) * 2));
            }
            for (int i = tid; i < 32 * 64; i += 256) {
                int kk = i >> 6, n = i & 63;
                sB[n * 40 + kk] = W1[(size_t)(k0 + kk) * 384 + n0 + n];
            }
            wait_async0();
            __syncthreads();
            v16bf af = ld_frag(sA + (wv * 16 + ln) * 40 + hg * 8,
                               sA + (wv * 16 + ln) * 40 + 16 + hg * 8);
#pragma unroll
            for (int nt = 0; nt < 4; ++nt) {
                v16bf bf = ld_frag(sB + (nt * 16 + ln) * 40 + hg * 8,
                                   sB + (nt * 16 + ln) * 40 + 16 + hg * 8);
                acc[nt] = wmma_bf16(af, bf, acc[nt]);
            }
            __syncthreads();
        }
#pragma unroll
        for (int nt = 0; nt < 4; ++nt)
#pragma unroll
            for (int v = 0; v < 8; ++v) {
                float val = acc[nt][v] + t2r[nt][v];
                val = (val > 0.0f) ? val : 0.2f * val;   // leaky relu
                if (val > best[nt][v]) best[nt][v] = val;
            }
    }
#pragma unroll
    for (int nt = 0; nt < 4; ++nt)
#pragma unroll
        for (int v = 0; v < 8; ++v) {
            int row = m0 + wv * 16 + hg * 8 + v, col = n0 + nt * 16 + ln;
            outB[(size_t)row * 768 + 384 + col] = f2bf(best[nt][v]);  // right half of cat
        }
}

// ---------------- host orchestration -----------------------------------------
extern "C" void kernel_launch(void* const* d_in, const int* in_sizes, int n_in,
                              void* d_out, int out_size, void* d_ws, size_t ws_size,
                              hipStream_t stream) {
    const float*     x        = (const float*)d_in[0];
    const long long* knn_idx  = (const long long*)d_in[1];
    const float*     norm1_w  = (const float*)d_in[2];
    const float*     norm1_b  = (const float*)d_in[3];
    const float*     qkv_w    = (const float*)d_in[4];
    const float*     proj_w   = (const float*)d_in[5];
    const float*     proj_b   = (const float*)d_in[6];
    const float*     knn_w    = (const float*)d_in[7];
    const float*     knn_b    = (const float*)d_in[8];
    const float*     merge_w  = (const float*)d_in[9];
    const float*     merge_b  = (const float*)d_in[10];
    const float*     norm2_w  = (const float*)d_in[11];
    const float*     norm2_b  = (const float*)d_in[12];
    const float*     fc1_w    = (const float*)d_in[13];
    const float*     fc1_b    = (const float*)d_in[14];
    const float*     fc2_w    = (const float*)d_in[15];
    const float*     fc2_b    = (const float*)d_in[16];
    float* out = (float*)d_out;

    const size_t M = 16384;  // B*N
    char* ws = (char*)d_ws;
    size_t off = 0;
    auto take = [&](size_t bytes) { char* p = ws + off; off = (off + bytes + 255) & ~(size_t)255; return p; };

    u16*   nxb   = (u16*)  take(M * 384 * 2);
    u16*   qkvb  = (u16*)  take((size_t)3 * 8 * 6 * 2048 * 64 * 2);
    u16*   aout  = (u16*)  take(M * 384 * 2);
    u16*   cat1  = (u16*)  take(M * 768 * 2);
    float* t2    = (float*)take(M * 384 * 4);
    float* y     = (float*)take(M * 384 * 4);
    u16*   h2    = (u16*)  take(M * 384 * 2);
    u16*   hid   = (u16*)  take(M * 1536 * 2);
    u16*   wqkv  = (u16*)  take((size_t)384 * 1152 * 2);
    u16*   wproj = (u16*)  take((size_t)384 * 384 * 2);
    u16*   w1    = (u16*)  take((size_t)384 * 384 * 2);
    u16*   wd    = (u16*)  take((size_t)384 * 384 * 2);
    u16*   wmrg  = (u16*)  take((size_t)768 * 384 * 2);
    u16*   wfc1  = (u16*)  take((size_t)384 * 1536 * 2);
    u16*   wfc2  = (u16*)  take((size_t)1536 * 384 * 2);

    auto cvt = [&](const float* src, u16* dst, int n) {
        cvt_k<<<(n + 255) / 256, 256, 0, stream>>>(src, dst, n);
    };
    cvt(qkv_w, wqkv, 384 * 1152);
    cvt(proj_w, wproj, 384 * 384);
    cvt(merge_w, wmrg, 768 * 384);
    cvt(fc1_w, wfc1, 384 * 1536);
    cvt(fc2_w, wfc2, 1536 * 384);
    cvt_wd_k<<<(384 * 384 + 255) / 256, 256, 0, stream>>>(knn_w, w1, wd, 384 * 384);

    // norm1
    ln_k<<<M, 128, 0, stream>>>(x, norm1_w, norm1_b, nxb, 384);

    // QKV GEMM with scatter epilogue (Q pre-scaled)
    gemm_bf16_k<2><<<dim3(1152 / 64, M / 128), 256, 0, stream>>>(
        nxb, 384, wqkv, 1152, M, 1152, 384, nullptr, nullptr, 0, nullptr, qkvb, 0, 0);

    // flash attention
    attn_k<<<8 * 6 * 32, 128, 0, stream>>>(qkvb, aout);

    // proj -> left half of concat buffer (bf16, stride 768)
    gemm_bf16_k<1><<<dim3(6, M / 128), 256, 0, stream>>>(
        aout, 384, wproj, 384, M, 384, 384, proj_b, nullptr, 0, nullptr, cat1, 768, 0);

    // t2 = norm_x @ (W2 - W1) + knn_b  (f32)
    gemm_bf16_k<0><<<dim3(6, M / 128), 256, 0, stream>>>(
        nxb, 384, wd, 384, M, 384, 384, knn_b, nullptr, 0, t2, nullptr, 384, 0);

    // KNN branch -> right half of concat buffer
    knn_k<<<dim3(6, M / 128), 256, 0, stream>>>(nxb, w1, knn_idx, t2, cat1);

    // merge GEMM + residual x -> y (f32)
    gemm_bf16_k<0><<<dim3(6, M / 128), 256, 0, stream>>>(
        cat1, 768, wmrg, 384, M, 384, 768, merge_b, x, 384, y, nullptr, 384, 0);

    // norm2
    ln_k<<<M, 128, 0, stream>>>(y, norm2_w, norm2_b, h2, 384);

    // fc1 + exact GELU (bf16 out)
    gemm_bf16_k<1><<<dim3(1536 / 64, M / 128), 256, 0, stream>>>(
        h2, 384, wfc1, 1536, M, 1536, 384, fc1_b, nullptr, 0, nullptr, hid, 1536, 1);

    // fc2 + residual y -> d_out (f32)
    gemm_bf16_k<0><<<dim3(6, M / 128), 256, 0, stream>>>(
        hid, 1536, wfc2, 384, M, 384, 1536, fc2_b, y, 384, out, nullptr, 384, 0);
}
